// EnhancedNASTransformer_79345225826823
// MI455X (gfx1250) — matrix operand order repaired and checked by reading, not compiled, not run on hardware
//
#include <hip/hip_runtime.h>
#include <math.h>

// ---------------- problem constants ----------------
static constexpr int B  = 2;
static constexpr int S  = 48;
static constexpr int E  = 512;
static constexpr int P  = 256;
static constexpr int H  = 8;
static constexpr int F  = 1024;
static constexpr int HDE = E / H;   // 64
static constexpr int HDP = P / H;   // 32
static constexpr int BS  = B * S;       // 96 msa tokens
static constexpr int TT  = S * S;       // 2304 flattened pair tokens per batch
static constexpr int BTT = B * TT;      // 4608
static constexpr int MSA_N  = B * S * E;        // 49152
static constexpr int PAIR_N = B * S * S * P;    // 1179648

// ---------------- WMMA plumbing (gfx1250, wave32) ----------------
typedef __bf16 bhalf;
typedef bhalf bhalf4  __attribute__((ext_vector_type(4)));
typedef bhalf bhalf16 __attribute__((ext_vector_type(16)));
typedef float floatx8 __attribute__((ext_vector_type(8)));

__device__ __forceinline__ floatx8 wmma_bf16(bhalf16 a, bhalf16 b, floatx8 c) {
  // D(16x16,f32) = A(16x32,bf16) * B(32x16,bf16) + C
  return __builtin_amdgcn_wmma_f32_16x16x32_bf16(false, a, false, b, (short)0, c, false, false);
}
// Fragment index helpers (ISA 7.12.2):
//  A 16x32 bf16 : lane l elem t -> M = l&15, K = (t&7) + ((t>>3)<<4) + ((l>>4)<<3)
//  B 32x16 bf16 : lane l elem t -> N = l&15, K = (t&7) + ((t>>3)<<4) + ((l>>4)<<3)
//  C/D 16x16 f32: lane l elem j -> N = l&15, M = j + ((l>>4)<<3)
#define A_KIDX(t, lane) (((t) & 7) + (((t) >> 3) << 4) + (((lane) >> 4) << 3))
#define B_KIDX(t, lane) (((t) & 7) + (((t) >> 3) << 4) + (((lane) >> 4) << 3))
#define C_MROW(j, lane) ((j) + (((lane) >> 4) << 3))

// Vectorized fragment loaders. Per-lane base must already include the lane's
// K sub-offset: A base += (lane>>4)<<3 ; B base += (lane>>4)<<3. 16B aligned.
__device__ __forceinline__ bhalf16 load_a_frag(const float* __restrict__ p) {
  float4 x0 = *(const float4*)(p);       // K+0..3
  float4 x1 = *(const float4*)(p + 4);   // K+4..7
  float4 x2 = *(const float4*)(p + 16);  // K+16..19
  float4 x3 = *(const float4*)(p + 20);  // K+20..23
  bhalf16 a;
  a[0]=(bhalf)x0.x; a[1]=(bhalf)x0.y; a[2]=(bhalf)x0.z; a[3]=(bhalf)x0.w;
  a[4]=(bhalf)x1.x; a[5]=(bhalf)x1.y; a[6]=(bhalf)x1.z; a[7]=(bhalf)x1.w;
  a[8]=(bhalf)x2.x; a[9]=(bhalf)x2.y; a[10]=(bhalf)x2.z; a[11]=(bhalf)x2.w;
  a[12]=(bhalf)x3.x; a[13]=(bhalf)x3.y; a[14]=(bhalf)x3.z; a[15]=(bhalf)x3.w;
  return a;
}
__device__ __forceinline__ bhalf16 load_a_frag_prod(const float* __restrict__ pi,
                                                    const float* __restrict__ pj) {
  float4 y0 = *(const float4*)(pi);      float4 z0 = *(const float4*)(pj);
  float4 y1 = *(const float4*)(pi + 4);  float4 z1 = *(const float4*)(pj + 4);
  float4 y2 = *(const float4*)(pi + 16); float4 z2 = *(const float4*)(pj + 16);
  float4 y3 = *(const float4*)(pi + 20); float4 z3 = *(const float4*)(pj + 20);
  bhalf16 a;
  a[0]=(bhalf)(y0.x*z0.x); a[1]=(bhalf)(y0.y*z0.y); a[2]=(bhalf)(y0.z*z0.z); a[3]=(bhalf)(y0.w*z0.w);
  a[4]=(bhalf)(y1.x*z1.x); a[5]=(bhalf)(y1.y*z1.y); a[6]=(bhalf)(y1.z*z1.z); a[7]=(bhalf)(y1.w*z1.w);
  a[8]=(bhalf)(y2.x*z2.x); a[9]=(bhalf)(y2.y*z2.y); a[10]=(bhalf)(y2.z*z2.z); a[11]=(bhalf)(y2.w*z2.w);
  a[12]=(bhalf)(y3.x*z3.x); a[13]=(bhalf)(y3.y*z3.y); a[14]=(bhalf)(y3.z*z3.z); a[15]=(bhalf)(y3.w*z3.w);
  return a;
}
__device__ __forceinline__ bhalf16 load_b_frag(const float* __restrict__ p) {
  float4 x0 = *(const float4*)(p);
  float4 x1 = *(const float4*)(p + 4);
  float4 x2 = *(const float4*)(p + 16);
  float4 x3 = *(const float4*)(p + 20);
  bhalf16 b;
  b[0]=(bhalf)x0.x; b[1]=(bhalf)x0.y; b[2]=(bhalf)x0.z; b[3]=(bhalf)x0.w;
  b[4]=(bhalf)x1.x; b[5]=(bhalf)x1.y; b[6]=(bhalf)x1.z; b[7]=(bhalf)x1.w;
  b[8]=(bhalf)x2.x; b[9]=(bhalf)x2.y; b[10]=(bhalf)x2.z; b[11]=(bhalf)x2.w;
  b[12]=(bhalf)x3.x; b[13]=(bhalf)x3.y; b[14]=(bhalf)x3.z; b[15]=(bhalf)x3.w;
  return b;
}
// Pre-packed bf16 weights: bf16 K+0..7 and K+16..23 (16B-aligned) -> 2x b128.
__device__ __forceinline__ bhalf16 load_b_frag_bf(const bhalf* __restrict__ p) {
  typedef bhalf bhalf8v __attribute__((ext_vector_type(8)));
  bhalf8v lo = *(const bhalf8v*)p;
  bhalf8v hi = *(const bhalf8v*)(p + 16);
  return __builtin_shufflevector(lo, hi, 0, 1, 2, 3, 4, 5, 6, 7, 8, 9, 10, 11, 12, 13, 14, 15);
}

__device__ __forceinline__ float gelu_exact(float x) {
  return 0.5f * x * (1.f + erff(x * 0.70710678118654752f));
}
__device__ __forceinline__ float sigmoid_fast(float x) {
  return __builtin_amdgcn_rcpf(1.f + __expf(-x));   // v_rcp_f32
}

// ---------------- one-time f32 -> bf16 weight pre-pack ----------------
__global__ void pack_bf16_kernel(const float* __restrict__ src, bhalf* __restrict__ dst, int n) {
  int i = (blockIdx.x * 256 + threadIdx.x) * 4;
  if (i >= n) return;
  float4 x = *(const float4*)(src + i);
  bhalf4 y;
  y[0] = (bhalf)x.x; y[1] = (bhalf)x.y; y[2] = (bhalf)x.z; y[3] = (bhalf)x.w;
  *(bhalf4*)(dst + i) = y;
}

// ---- generic GEMM: C = act(A[M,K](f32) * W[N,K](bf16)^T + bias) (*mul) (+res) ----
// One wave per 16x64 output strip: 1 A-frag feeds 4 WMMAs per K-step.
template<int ACT, bool HAS_MUL, bool HAS_RES>
__global__ void gemm_kernel(const float* __restrict__ A, const bhalf* __restrict__ W,
                            const float* __restrict__ bias, const float* __restrict__ res,
                            const float* __restrict__ mul, float* __restrict__ C,
                            int M, int N, int K) {
  int lane = threadIdx.x;
  int tn = blockIdx.x * 64, tm = blockIdx.y * 16;
  int m  = tm + (lane & 15);
  int n0 = tn + (lane & 15);
  const float* ar = A + (size_t)m * K + ((lane >> 4) << 3);
  const bhalf* wr = W + (size_t)n0 * K + ((lane >> 4) << 3);
  floatx8 acc[4] = {};
  for (int k0 = 0; k0 < K; k0 += 32) {
    bhalf16 af = load_a_frag(ar + k0);
#pragma unroll
    for (int nt = 0; nt < 4; ++nt)
      acc[nt] = wmma_bf16(af, load_b_frag_bf(wr + (size_t)(nt * 16) * K + k0), acc[nt]);
  }
#pragma unroll
  for (int nt = 0; nt < 4; ++nt) {
    int n = n0 + nt * 16;
    float bv = bias[n];
#pragma unroll
    for (int j = 0; j < 8; ++j) {
      int mm = tm + C_MROW(j, lane);
      float v = acc[nt][j] + bv;
      if constexpr (ACT == 1) v = gelu_exact(v);
      if constexpr (ACT == 2) v = sigmoid_fast(v);
      size_t idx = (size_t)mm * N + n;
      if constexpr (HAS_MUL) v *= mul[idx];
      if constexpr (HAS_RES) v += res[idx];
      C[idx] = v;
    }
  }
}

static void launch_gemm(const float* A, const bhalf* W_, const float* b_,
                        const float* res, const float* mul, float* C,
                        int M, int N, int K, int act, hipStream_t stream) {
  dim3 g(N / 64, M / 16), blk(32);
  if (mul && res)
    gemm_kernel<0, true, true><<<g, blk, 0, stream>>>(A, W_, b_, res, mul, C, M, N, K);
  else if (res)
    gemm_kernel<0, false, true><<<g, blk, 0, stream>>>(A, W_, b_, res, mul, C, M, N, K);
  else if (act == 1)
    gemm_kernel<1, false, false><<<g, blk, 0, stream>>>(A, W_, b_, res, mul, C, M, N, K);
  else if (act == 2)
    gemm_kernel<2, false, false><<<g, blk, 0, stream>>>(A, W_, b_, res, mul, C, M, N, K);
  else
    gemm_kernel<0, false, false><<<g, blk, 0, stream>>>(A, W_, b_, res, mul, C, M, N, K);
}

// ---------------- LayerNorm over last dim C ----------------
__global__ void layernorm_kernel(const float* __restrict__ x, const float* __restrict__ g,
                                 const float* __restrict__ b, float* __restrict__ y, int C) {
  int t = blockIdx.x;
  const float* xr = x + (size_t)t * C;
  float* yr = y + (size_t)t * C;
  __shared__ float red[256];
  int tid = threadIdx.x;
  float s = 0.f;
  for (int c = tid; c < C; c += 256) s += xr[c];
  red[tid] = s; __syncthreads();
  for (int off = 128; off > 0; off >>= 1) { if (tid < off) red[tid] += red[tid + off]; __syncthreads(); }
  float mean = red[0] / C; __syncthreads();
  float var = 0.f;
  for (int c = tid; c < C; c += 256) { float d = xr[c] - mean; var += d * d; }
  red[tid] = var; __syncthreads();
  for (int off = 128; off > 0; off >>= 1) { if (tid < off) red[tid] += red[tid + off]; __syncthreads(); }
  float rstd = rsqrtf(red[0] / C + 1e-5f);
  for (int c = tid; c < C; c += 256) yr[c] = (xr[c] - mean) * rstd * g[c] + b[c];
}

// ---- pair bias: out[b,h,i,j] = pair[b,i,j,:] . w_pb[h,:] + b_pb[h] ----
__global__ void pair_bias_kernel(const float* __restrict__ pair, const float* __restrict__ w,
                                 const float* __restrict__ bb, float* __restrict__ out, int total) {
  int idx = blockIdx.x * 256 + threadIdx.x;
  if (idx >= total) return;
  int j = idx % S; int tmp = idx / S;
  int i = tmp % S; tmp /= S;
  int h = tmp % H; int b = tmp / H;
  const float4* pp = (const float4*)(pair + ((size_t)((b * S + i) * S + j)) * P);
  const float4* wr = (const float4*)(w + h * P);
  float s = bb[h];
  for (int c = 0; c < P / 4; ++c) {
    float4 a = pp[c], g = wr[c];
    s += a.x * g.x + a.y * g.y + a.z * g.z + a.w * g.w;
  }
  out[idx] = s;
}

// ---------------- MSA scores: sc[b,h,i,j] = q.k/8 + bias ----------------
__global__ void msa_scores_kernel(const float* __restrict__ q, const float* __restrict__ k,
                                  const float* __restrict__ bias, float* __restrict__ sc) {
  int lane = threadIdx.x;
  int tn = blockIdx.x * 16, tm = blockIdx.y * 16;
  int bh = blockIdx.z; int b = bh >> 3; int h = bh & 7;
  int i = tm + (lane & 15);
  int j = tn + (lane & 15);
  const float* qr = q + ((size_t)(b * S + i) * H + h) * HDE + ((lane >> 4) << 3);
  const float* kr = k + ((size_t)(b * S + j) * H + h) * HDE + ((lane >> 4) << 3);
  floatx8 acc = {};
  for (int k0 = 0; k0 < HDE; k0 += 32)
    acc = wmma_bf16(load_a_frag(qr + k0), load_b_frag(kr + k0), acc);
#pragma unroll
  for (int jj = 0; jj < 8; ++jj) {
    int ii = tm + C_MROW(jj, lane);
    size_t idx = ((size_t)bh * S + ii) * S + j;
    sc[idx] = acc[jj] * 0.125f + bias[idx];
  }
}

// ---------------- masked softmax over rows of 48 ----------------
__global__ void msa_softmax_kernel(float* __restrict__ sc, const unsigned char* __restrict__ mask) {
  int row = blockIdx.x;              // [B*H*S]
  int b = row / (H * S);
  __shared__ float red[64];
  int j = threadIdx.x;
  float* r = sc + (size_t)row * S;
  bool valid = (j < S) && (mask[b * S + j] != 0);
  float v = valid ? r[j] : -1e30f;
  red[j] = v; __syncthreads();
  for (int off = 32; off > 0; off >>= 1) { if (j < off) red[j] = fmaxf(red[j], red[j + off]); __syncthreads(); }
  float mx = red[0]; __syncthreads();
  float e = valid ? __expf(v - mx) : 0.f;
  red[j] = e; __syncthreads();
  for (int off = 32; off > 0; off >>= 1) { if (j < off) red[j] += red[j + off]; __syncthreads(); }
  float sum = red[0];
  float inv = (sum > 0.f) ? 1.f / sum : 0.f;
  if (j < S) r[j] = e * inv;
}

// ---------------- MSA PV: ctx[b,i,h,d] = sum_j attn * v, K = 48 (guarded) ------
__global__ void msa_pv_kernel(const float* __restrict__ attn, const float* __restrict__ v,
                              float* __restrict__ ctx) {
  int lane = threadIdx.x;
  int tn = blockIdx.x * 16, tm = blockIdx.y * 16;
  int bh = blockIdx.z; int b = bh >> 3; int h = bh & 7;
  int i = tm + (lane & 15);
  int d = tn + (lane & 15);
  floatx8 acc = {};
  for (int k0 = 0; k0 < S; k0 += 32) {
    bhalf16 af, bf;
#pragma unroll
    for (int t = 0; t < 16; ++t) {
      int ka = k0 + A_KIDX(t, lane);
      af[t] = (bhalf)((ka < S) ? attn[((size_t)bh * S + i) * S + ka] : 0.f);
      int kb = k0 + B_KIDX(t, lane);
      bf[t] = (bhalf)((kb < S) ? v[((size_t)(b * S + kb) * H + h) * HDE + d] : 0.f);
    }
    acc = wmma_bf16(af, bf, acc);
  }
#pragma unroll
  for (int jj = 0; jj < 8; ++jj) {
    int ii = tm + C_MROW(jj, lane);
    ctx[((size_t)(b * S + ii) * H + h) * HDE + d] = acc[jj];
  }
}

// ------- fused outer-product pair update (16x64 strip per wave) -------
// pair[b,i,j,:] += ((rp_i * rp_j) W^T + b) * pmf
__global__ void r2p_kernel(const float* __restrict__ rp, const bhalf* __restrict__ W,
                           const float* __restrict__ bias, const unsigned char* __restrict__ mask,
                           float* __restrict__ pair) {
  int lane = threadIdx.x;
  int tn = blockIdx.x * 64, tm = blockIdx.y * 16;
  int tok = tm + (lane & 15);
  int b = tok / TT; int rem = tok - b * TT; int i = rem / S; int j = rem - i * S;
  const float* rpi = rp + (size_t)(b * S + i) * E + ((lane >> 4) << 3);
  const float* rpj = rp + (size_t)(b * S + j) * E + ((lane >> 4) << 3);
  int n0 = tn + (lane & 15);
  const bhalf* wr = W + (size_t)n0 * E + ((lane >> 4) << 3);
  floatx8 acc[4] = {};
  for (int k0 = 0; k0 < E; k0 += 32) {
    bhalf16 af = load_a_frag_prod(rpi + k0, rpj + k0);
#pragma unroll
    for (int nt = 0; nt < 4; ++nt)
      acc[nt] = wmma_bf16(af, load_b_frag_bf(wr + (size_t)(nt * 16) * E + k0), acc[nt]);
  }
#pragma unroll
  for (int nt = 0; nt < 4; ++nt) {
    int n = n0 + nt * 16;
    float bv = bias[n];
#pragma unroll
    for (int jj = 0; jj < 8; ++jj) {
      int mm = tm + C_MROW(jj, lane);
      int bb = mm / TT; int r2 = mm - bb * TT; int ii = r2 / S; int j2 = r2 - ii * S;
      float pmf = (mask[bb * S + ii] && mask[bb * S + j2]) ? 1.f : 0.f;
      size_t idx = (size_t)mm * P + n;
      pair[idx] += (acc[nt][jj] + bv) * pmf;
    }
  }
}

// ------- triangle einsum, batched over (b,c): prod[i,j,c] = sum_k l.r  (*pmf) ----
__global__ void tri_kernel(const float* __restrict__ L, const float* __restrict__ R,
                           const unsigned char* __restrict__ mask, float* __restrict__ prod,
                           int outgoing) {
  int lane = threadIdx.x;
  int tn = blockIdx.x * 16, tm = blockIdx.y * 16;
  int z = blockIdx.z; int b = z >> 8; int c = z & 255;
  int i = tm + (lane & 15);
  int j = tn + (lane & 15);
  floatx8 acc = {};
  for (int k0 = 0; k0 < S; k0 += 32) {
    bhalf16 af, bf;
#pragma unroll
    for (int t = 0; t < 16; ++t) {
      int ka = k0 + A_KIDX(t, lane);
      float av = 0.f;
      if (ka < S)
        av = outgoing ? L[((size_t)(b * S + i) * S + ka) * P + c]
                      : L[((size_t)(b * S + ka) * S + i) * P + c];
      af[t] = (bhalf)av;
      int kb = k0 + B_KIDX(t, lane);
      float bv = 0.f;
      if (kb < S)
        bv = outgoing ? R[((size_t)(b * S + j) * S + kb) * P + c]
                      : R[((size_t)(b * S + kb) * S + j) * P + c];
      bf[t] = (bhalf)bv;
    }
    acc = wmma_bf16(af, bf, acc);
  }
#pragma unroll
  for (int jj = 0; jj < 8; ++jj) {
    int ii = tm + C_MROW(jj, lane);
    float pmf = (mask[b * S + ii] && mask[b * S + j]) ? 1.f : 0.f;
    prod[((size_t)(b * S + ii) * S + j) * P + c] = acc[jj] * pmf;
  }
}

// ------- flash-style pairwise attention over 2304 tokens, per (b,h,i-tile) -------
__global__ void pair_attn_kernel(const float* __restrict__ q, const float* __restrict__ k,
                                 const float* __restrict__ v, const unsigned char* __restrict__ mask,
                                 float* __restrict__ ctx) {
  int lane = threadIdx.x;
  int it = blockIdx.x;               // 0..143 (i-tile of 16 tokens)
  int bh = blockIdx.y; int b = bh >> 3; int h = bh & 7;
  __shared__ float sprob[16][40];    // stride 40 -> 16B-aligned rows for b128 LDS reads
  __shared__ float srow[16];
  int mloc = lane & 15;
  int hi = lane >> 4;
  int ti = it * 16 + mloc;
  // Q fragment: A 16x32 (rows = i tokens, K = head dim 32)
  bhalf16 aq = load_a_frag(q + ((size_t)b * TT + ti) * P + h * HDP + (hi << 3));
  const float* kcol = k + ((size_t)b * TT + mloc) * P + h * HDP + (hi << 3);
  float rmax = -1e30f, rl = 0.f;     // per-row state (meaningful in lanes 0..15)
  floatx8 o1 = {}, o2 = {};
  const float scl = 0.17677669529663687f;  // 1/sqrt(32)
  for (int j0 = 0; j0 < TT; j0 += 32) {
    bhalf16 bk1 = load_b_frag(kcol + (size_t)j0 * P);
    bhalf16 bk2 = load_b_frag(kcol + (size_t)(j0 + 16) * P);
    floatx8 s1 = {}, s2 = {};
    s1 = wmma_bf16(aq, bk1, s1);
    s2 = wmma_bf16(aq, bk2, s2);
#pragma unroll
    for (int jj = 0; jj < 8; ++jj) {
      int mm = C_MROW(jj, lane);
      sprob[mm][mloc]      = s1[jj] * scl;
      sprob[mm][16 + mloc] = s2[jj] * scl;
    }
    __syncthreads();
    if (lane < 16) {                 // online softmax for row `lane`
      float mx = rmax;
      for (int jj = 0; jj < 32; ++jj) {
        int tj = j0 + jj;
        int ji = tj / S, j2 = tj - ji * S;
        bool valid = mask[b * S + ji] && mask[b * S + j2];
        float sv = valid ? sprob[lane][jj] : -1e30f;
        sprob[lane][jj] = sv;
        mx = fmaxf(mx, sv);
      }
      float scale = __expf(rmax - mx);
      float ls = rl * scale;
      for (int jj = 0; jj < 32; ++jj) {
        float sv = sprob[lane][jj];
        float pp = (sv <= -1e29f) ? 0.f : __expf(sv - mx);
        ls += pp;
        sprob[lane][jj] = pp;
      }
      rmax = mx; rl = ls;
      srow[lane] = scale;
    }
    __syncthreads();
#pragma unroll
    for (int jj = 0; jj < 8; ++jj) {  // rescale accumulator rows
      float f = srow[C_MROW(jj, lane)];
      o1[jj] *= f; o2[jj] *= f;
    }
    bhalf16 ap = load_a_frag(&sprob[mloc][hi << 3]);   // probs from LDS (b128)
    bhalf16 bv1, bv2;
#pragma unroll
    for (int t = 0; t < 16; ++t) {                     // V rows are strided by P
      int jj = B_KIDX(t, lane);
      size_t base = ((size_t)b * TT + j0 + jj) * P + h * HDP;
      bv1[t] = (bhalf)v[base + mloc];
      bv2[t] = (bhalf)v[base + 16 + mloc];
    }
    o1 = wmma_bf16(ap, bv1, o1);
    o2 = wmma_bf16(ap, bv2, o2);
    __syncthreads();
  }
  if (lane < 16) srow[lane] = (rl > 0.f) ? 1.f / rl : 0.f;
  __syncthreads();
#pragma unroll
  for (int jj = 0; jj < 8; ++jj) {
    int mm = C_MROW(jj, lane);
    float f = srow[mm];
    size_t base = ((size_t)b * TT + it * 16 + mm) * P + h * HDP + mloc;
    ctx[base]      = o1[jj] * f;
    ctx[base + 16] = o2[jj] * f;
  }
}

// ---------------- host orchestration ----------------
// params leaf order (insertion order of setup_inputs):
// 3..10 ra_{q,k,v,o}(w,b) | 11,12 ra_pb | 13,14 ra_ln | 15,16 rff_ln | 17,18 rff_w1 |
// 19,20 rff_w2 | 21,22 r2p_ln(unused) | 23,24 r2p_row | 25,26 r2p_out |
// 27,28 to_ln | 29..36 to_{l,r,g,o} | 37,38 ti_ln | 39..46 ti_{l,r,g,o} |
// 47..54 pa_{q,k,v,o} | 55,56 pa_ln | 57,58 pff_ln | 59,60 pff_w1 | 61,62 pff_w2

extern "C" void kernel_launch(void* const* d_in, const int* in_sizes, int n_in,
                              void* d_out, int out_size, void* d_ws, size_t ws_size,
                              hipStream_t stream) {
  (void)in_sizes; (void)n_in; (void)out_size; (void)ws_size;
  const unsigned char* mask = (const unsigned char*)d_in[2];
  auto PW = [&](int i) { return (const float*)d_in[i]; };

  float* msa0  = (float*)d_out;           // accumulated msa  [B,S,E]
  float* pair0 = msa0 + MSA_N;            // accumulated pair [B,S,S,P]

  float* w = (float*)d_ws;
  float* t0 = w; w += MSA_N;              // msa LN out
  float* t1 = w; w += MSA_N;              // q / rp
  float* t2 = w; w += MSA_N;              // k
  float* t3 = w; w += MSA_N;              // v
  float* t4 = w; w += MSA_N;              // ctx
  float* tb = w; w += B * H * S * S;      // pair bias
  float* ts = w; w += B * H * S * S;      // msa scores
  float* th = w; w += B * S * F;          // msa FF hidden
  float* p1 = w; w += PAIR_N;             // pair LN out
  float* p2 = w; w += PAIR_N;             // l / q
  float* p3 = w; w += PAIR_N;             // r / k
  float* p4 = w; w += PAIR_N;             // prod / v
  float* p5 = w; w += PAIR_N;             // gate / ctx
  float* pf = w; w += (size_t)BTT * F;    // pair FF hidden

  hipMemcpyAsync(msa0,  d_in[0], (size_t)MSA_N  * sizeof(float), hipMemcpyDeviceToDevice, stream);
  hipMemcpyAsync(pair0, d_in[1], (size_t)PAIR_N * sizeof(float), hipMemcpyDeviceToDevice, stream);

  // ---- one-time weight pre-pack to bf16 (rides in workspace after fp32 区) ----
  static const struct { int idx; int n; } packs[] = {
    {3, E*E}, {5, E*E}, {7, E*E}, {9, E*E}, {17, F*E}, {19, E*F}, {23, E*E}, {25, P*E},
    {29, P*P}, {31, P*P}, {33, P*P}, {35, P*P}, {39, P*P}, {41, P*P}, {43, P*P}, {45, P*P},
    {47, P*P}, {49, P*P}, {51, P*P}, {53, P*P}, {59, F*P}, {61, P*F}
  };
  const bhalf* WB[64] = {};
  bhalf* wb = (bhalf*)w;
  for (auto& pk : packs) {
    pack_bf16_kernel<<<pk.n / 1024, 256, 0, stream>>>(PW(pk.idx), wb, pk.n);
    WB[pk.idx] = wb;
    wb += pk.n;
  }

  // gemm(A, weight-idx, res, mul, C, M, N, K, act): bias is always idx+1
  auto gemm = [&](const float* A, int wi, const float* res, const float* mul, float* C,
                  int M, int N, int K, int act) {
    launch_gemm(A, WB[wi], PW(wi + 1), res, mul, C, M, N, K, act, stream);
  };

  // ===== MSA row attention with pair bias =====
  layernorm_kernel<<<BS, 256, 0, stream>>>(msa0, PW(13), PW(14), t0, E);
  gemm(t0, 3, nullptr, nullptr, t1, BS, E, E, 0);   // q
  gemm(t0, 5, nullptr, nullptr, t2, BS, E, E, 0);   // k
  gemm(t0, 7, nullptr, nullptr, t3, BS, E, E, 0);   // v
  pair_bias_kernel<<<(B * H * S * S + 255) / 256, 256, 0, stream>>>(
      (const float*)d_in[1], PW(11), PW(12), tb, B * H * S * S);
  msa_scores_kernel<<<dim3(S / 16, S / 16, B * H), dim3(32), 0, stream>>>(t1, t2, tb, ts);
  msa_softmax_kernel<<<B * H * S, 64, 0, stream>>>(ts, mask);
  msa_pv_kernel<<<dim3(HDE / 16, S / 16, B * H), dim3(32), 0, stream>>>(ts, t3, t4);
  gemm(t4, 9, msa0, nullptr, msa0, BS, E, E, 0);    // o-proj + residual

  // ===== MSA FF =====
  layernorm_kernel<<<BS, 256, 0, stream>>>(msa0, PW(15), PW(16), t0, E);
  gemm(t0, 17, nullptr, nullptr, th, BS, F, E, 1);  // gelu
  gemm(th, 19, msa0, nullptr, msa0, BS, E, F, 0);

  // ===== row -> pair outer product update =====
  gemm(msa0, 23, nullptr, nullptr, t1, BS, E, E, 0);   // rp
  r2p_kernel<<<dim3(P / 64, BTT / 16), dim3(32), 0, stream>>>(t1, WB[25], PW(26), mask, pair0);

  // ===== triangle multiplications =====
  auto triangle = [&](int lnG, int lW, int rW, int gW, int oW, int outgoing) {
    layernorm_kernel<<<BTT, 256, 0, stream>>>(pair0, PW(lnG), PW(lnG + 1), p1, P);
    gemm(p1, lW, nullptr, nullptr, p2, BTT, P, P, 0);
    gemm(p1, rW, nullptr, nullptr, p3, BTT, P, P, 0);
    tri_kernel<<<dim3(S / 16, S / 16, B * P), dim3(32), 0, stream>>>(p2, p3, mask, p4, outgoing);
    gemm(p1, gW, nullptr, nullptr, p5, BTT, P, P, 2);       // sigmoid gate
    gemm(p4, oW, pair0, p5, pair0, BTT, P, P, 0);           // gate*o + residual
  };
  triangle(27, 29, 31, 33, 35, 1);   // outgoing
  triangle(37, 39, 41, 43, 45, 0);   // incoming

  // ===== pairwise self-attention over S*S tokens (flash) =====
  layernorm_kernel<<<BTT, 256, 0, stream>>>(pair0, PW(55), PW(56), p1, P);
  gemm(p1, 47, nullptr, nullptr, p2, BTT, P, P, 0);  // q
  gemm(p1, 49, nullptr, nullptr, p3, BTT, P, P, 0);  // k
  gemm(p1, 51, nullptr, nullptr, p4, BTT, P, P, 0);  // v
  pair_attn_kernel<<<dim3(TT / 16, B * H), dim3(32), 0, stream>>>(p2, p3, p4, mask, p5);
  gemm(p5, 53, pair0, nullptr, pair0, BTT, P, P, 0); // o-proj + residual

  // ===== pair FF =====
  layernorm_kernel<<<BTT, 256, 0, stream>>>(pair0, PW(57), PW(58), p1, P);
  gemm(p1, 59, nullptr, nullptr, pf, BTT, F, P, 1);  // gelu
  gemm(pf, 61, pair0, nullptr, pair0, BTT, P, F, 0);
}
